// CausalSelfAttention_43035572306229
// MI455X (gfx1250) — compile-verified, matrix operand order checked by reading, not run on hardware
//
#include <hip/hip_runtime.h>
#include <hip/hip_bf16.h>
#include <stdint.h>

// ---------------------------------------------------------------------------
// CDNA5 (gfx1250) causal self-attention, bf16 WMMA (v_wmma_f32_16x16x32_bf16)
// - weights transposed to N-major at conversion => all B fragments are two
//   contiguous 16B loads per lane
// - GEMMs: 256x64 block tile, 32x64 per wave, double-buffered LDS staging,
//   one barrier per 32-K step, B fragments preloaded into distinct regs
// - attention: flash-style, online softmax, score scale folded into Q
// ---------------------------------------------------------------------------

typedef __attribute__((ext_vector_type(16))) __bf16 v16bf;
typedef __attribute__((ext_vector_type(8)))  float  v8f;

#define Bn 2
#define Tn 2048
#define Cn 1024
#define Hn 16
#define Dn 64
#define C3 3072
#define Mrows 4096   // B*T

__device__ __forceinline__ unsigned short f2bf(float f) {
  unsigned int u = __float_as_uint(f);
  unsigned int r = (u + 0x7FFFu + ((u >> 16) & 1u)) >> 16;  // round-nearest-even
  return (unsigned short)r;
}

union FragB {            // one WMMA 16-bit operand: 8 VGPRs / 16 bf16 per lane
  v16bf v;
  uint4 q[2];
  unsigned short u[16];
};

// ---------------------------------------------------------------------------
// fp32 -> bf16 bulk convert (8 elements / thread, vectorized)
// ---------------------------------------------------------------------------
__global__ __launch_bounds__(256) void cvt_bf16(const float* __restrict__ src,
                                                unsigned short* __restrict__ dst,
                                                int n) {
  int i = (blockIdx.x * 256 + threadIdx.x) * 8;
  if (i >= n) return;
  const float4* s4 = (const float4*)(src + i);
  float4 a = s4[0], b = s4[1];
  union { unsigned short u[8]; uint4 q; } o;
  o.u[0] = f2bf(a.x); o.u[1] = f2bf(a.y); o.u[2] = f2bf(a.z); o.u[3] = f2bf(a.w);
  o.u[4] = f2bf(b.x); o.u[5] = f2bf(b.y); o.u[6] = f2bf(b.z); o.u[7] = f2bf(b.w);
  *(uint4*)(dst + i) = o.q;
}

// ---------------------------------------------------------------------------
// fp32 [K][N] -> bf16 [N][K] transpose-convert, 32x32 LDS tiles
// ---------------------------------------------------------------------------
__global__ __launch_bounds__(256) void cvt_t_bf16(const float* __restrict__ src,
                                                  unsigned short* __restrict__ dst,
                                                  int K, int N) {
  __shared__ float tile[32][33];
  const int tx = threadIdx.x & 31;
  const int ty = threadIdx.x >> 5;          // 8 rows per pass
  const int k0 = blockIdx.y * 32, n0 = blockIdx.x * 32;
  #pragma unroll
  for (int p = 0; p < 4; ++p) {
    int kr = ty + p * 8;
    tile[kr][tx] = src[(size_t)(k0 + kr) * N + n0 + tx];
  }
  __syncthreads();
  #pragma unroll
  for (int p = 0; p < 4; ++p) {
    int nr = ty + p * 8;
    dst[(size_t)(n0 + nr) * K + k0 + tx] = f2bf(tile[tx][nr]);
  }
}

// ---------------------------------------------------------------------------
// Shared GEMM core: acc[2][4] (32M x 64N per wave), double-buffered B tile.
// A: [Mrows][Cn] bf16 row-major.  BT: [N][Cn] bf16 (N-major).
// ---------------------------------------------------------------------------
__device__ __forceinline__ void gemm_core(const unsigned short* __restrict__ A,
                                          const unsigned short* __restrict__ BT,
                                          unsigned short (*Bt)[64][40],
                                          int mBase, int nBase,
                                          int ln, int ko, int sn, int sseg,
                                          v8f acc[2][4]) {
  const size_t arow0 = (size_t)(mBase + ln) * Cn;
  const size_t arow1 = arow0 + (size_t)16 * Cn;
  const unsigned short* bsrc = BT + (size_t)(nBase + sn) * Cn + sseg;

  // prologue: stage buffer 0
  *(uint4*)&Bt[0][sn][sseg] = *(const uint4*)(bsrc);

  for (int kb = 0; kb < Cn; kb += 32) {
    const int cur = (kb >> 5) & 1;
    __syncthreads();                       // staging of `cur` visible

    // kick off next tile's global load early
    uint4 nx;
    const bool have_next = (kb + 32 < Cn);
    if (have_next) nx = *(const uint4*)(bsrc + kb + 32);

    // A fragments (two 16-row halves)
    FragB a0, a1;
    a0.q[0] = *(const uint4*)(A + arow0 + kb + ko);
    a0.q[1] = *(const uint4*)(A + arow0 + kb + 16 + ko);
    a1.q[0] = *(const uint4*)(A + arow1 + kb + ko);
    a1.q[1] = *(const uint4*)(A + arow1 + kb + 16 + ko);

    // all 4 B fragments into distinct registers, then WMMA burst
    FragB bf[4];
    #pragma unroll
    for (int nt = 0; nt < 4; ++nt) {
      bf[nt].q[0] = *(const uint4*)&Bt[cur][nt * 16 + ln][ko];
      bf[nt].q[1] = *(const uint4*)&Bt[cur][nt * 16 + ln][16 + ko];
    }
    #pragma unroll
    for (int nt = 0; nt < 4; ++nt) {
      acc[0][nt] = __builtin_amdgcn_wmma_f32_16x16x32_bf16(
          false, a0.v, false, bf[nt].v, (short)0, acc[0][nt], false, false);
      acc[1][nt] = __builtin_amdgcn_wmma_f32_16x16x32_bf16(
          false, a1.v, false, bf[nt].v, (short)0, acc[1][nt], false, false);
    }

    // store next tile into the other buffer (reused 2 steps later; the single
    // barrier per step provides the needed ordering)
    if (have_next) *(uint4*)&Bt[cur ^ 1][sn][sseg] = nx;
  }
}

// ---------------------------------------------------------------------------
// GEMM1: qkv = Xb @ WabT, scatter into Q (pre-scaled by 1/sqrt(D)),
// K [B,H,T,D] and V transposed [B,H,D,T].
// ---------------------------------------------------------------------------
__global__ __launch_bounds__(256) void gemm_qkv(const unsigned short* __restrict__ Xb,
                                                const unsigned short* __restrict__ WabT, // [3072][1024]
                                                unsigned short* __restrict__ Qm,
                                                unsigned short* __restrict__ Km,
                                                unsigned short* __restrict__ Vt) {
  __shared__ __align__(16) unsigned short Bt[2][64][40];
  const int wave = threadIdx.x >> 5, lane = threadIdx.x & 31;
  const int hi = lane >> 4, ln = lane & 15, ko = hi * 8;
  const int mBase = blockIdx.y * 256 + wave * 32;
  const int nBase = blockIdx.x * 64;
  const int sn = threadIdx.x >> 2, sseg = (threadIdx.x & 3) * 8;

  v8f acc[2][4] = {};
  gemm_core(Xb, WabT, Bt, mBase, nBase, ln, ko, sn, sseg, acc);

  const int sec = nBase >> 10;               // 0=q, 1=k, 2=v
  const float sc = (sec == 0) ? 0.125f : 1.0f;   // fold 1/sqrt(64) into Q
  #pragma unroll
  for (int mh = 0; mh < 2; ++mh) {
    #pragma unroll
    for (int nt = 0; nt < 4; ++nt) {
      #pragma unroll
      for (int r = 0; r < 8; ++r) {
        int gm = mBase + mh * 16 + r + 8 * hi;     // D-matrix: row = r + 8*hi
        int c3 = nBase + nt * 16 + ln;             //           col = lane%16
        int c  = c3 & (Cn - 1);
        int b  = gm >> 11, tt = gm & (Tn - 1);
        int h  = c >> 6,  d  = c & 63;
        unsigned short v = f2bf(acc[mh][nt][r] * sc);
        if (sec == 0)      Qm[(((size_t)(b * Hn + h) * Tn) + tt) * Dn + d] = v;
        else if (sec == 1) Km[(((size_t)(b * Hn + h) * Tn) + tt) * Dn + d] = v;
        else               Vt[(((size_t)(b * Hn + h) * Dn) + d) * Tn + tt] = v;
      }
    }
  }
}

// ---------------------------------------------------------------------------
// Flash attention: 4 independent waves per block, 1 wave per 16-query tile,
// key blocks of 32, online softmax.  S = Q K^T (4 WMMA), P via per-wave LDS
// slice to A-layout, Y += P V (4 WMMA).  No inter-wave sync needed.
// Q is pre-scaled, so scores need no multiply here.
// ---------------------------------------------------------------------------
__global__ __launch_bounds__(128) void attn(const unsigned short* __restrict__ Qm,
                                            const unsigned short* __restrict__ Km,
                                            const unsigned short* __restrict__ Vt,
                                            unsigned short* __restrict__ Yb) {
  __shared__ __align__(16) unsigned short Ps[4][16][40];   // per-wave 16x32 P tile
  const int wave = threadIdx.x >> 5;
  const int lane = threadIdx.x & 31;
  const int hi = lane >> 4, ln = lane & 15, ko = hi * 8;
  const int gtile = blockIdx.x * 4 + wave;   // global 16-query tile id
  const int qt = gtile & 127;                // T/16 = 128 query tiles
  const int h  = (gtile >> 7) & 15;
  const int b  = gtile >> 11;
  const int qb = qt * 16;

  const unsigned short* Qp = Qm + (size_t)(b * Hn + h) * Tn * Dn;
  const unsigned short* Kp = Km + (size_t)(b * Hn + h) * Tn * Dn;
  const unsigned short* Vp = Vt + (size_t)(b * Hn + h) * Dn * Tn;

  // Q A-fragments for d = 0..31 and 32..63
  FragB aq[2];
  {
    int q = qb + ln;
    aq[0].q[0] = *(const uint4*)(Qp + (size_t)q * Dn + ko);
    aq[0].q[1] = *(const uint4*)(Qp + (size_t)q * Dn + 16 + ko);
    aq[1].q[0] = *(const uint4*)(Qp + (size_t)q * Dn + 32 + ko);
    aq[1].q[1] = *(const uint4*)(Qp + (size_t)q * Dn + 48 + ko);
  }

  v8f yacc[4] = {};
  float mrow[8], lrow[8];
  #pragma unroll
  for (int r = 0; r < 8; ++r) { mrow[r] = -1e30f; lrow[r] = 0.0f; }

  for (int kb = 0; kb <= qb + 15; kb += 32) {
    // ---- scores: two 16x16 tiles (keys kb..kb+15, kb+16..kb+31) ----
    v8f s0 = {}, s1 = {};
    #pragma unroll
    for (int db = 0; db < 2; ++db) {
      FragB kf;
      const unsigned short* kr0 = Kp + (size_t)(kb + ln) * Dn + db * 32;
      kf.q[0] = *(const uint4*)(kr0 + ko);
      kf.q[1] = *(const uint4*)(kr0 + 16 + ko);
      s0 = __builtin_amdgcn_wmma_f32_16x16x32_bf16(
          false, aq[db].v, false, kf.v, (short)0, s0, false, false);
      const unsigned short* kr1 = Kp + (size_t)(kb + 16 + ln) * Dn + db * 32;
      kf.q[0] = *(const uint4*)(kr1 + ko);
      kf.q[1] = *(const uint4*)(kr1 + 16 + ko);
      s1 = __builtin_amdgcn_wmma_f32_16x16x32_bf16(
          false, aq[db].v, false, kf.v, (short)0, s1, false, false);
    }

    // ---- causal mask + online softmax (row m lives in one 16-lane half) ----
    #pragma unroll
    for (int r = 0; r < 8; ++r) {
      int m = r + 8 * hi;
      int q = qb + m;
      float v0 = (kb + ln      <= q) ? s0[r] : -1e30f;
      float v1 = (kb + 16 + ln <= q) ? s1[r] : -1e30f;
      float tm = fmaxf(v0, v1);
      #pragma unroll
      for (int msk = 1; msk < 16; msk <<= 1) tm = fmaxf(tm, __shfl_xor(tm, msk, 32));
      float nm    = fmaxf(mrow[r], tm);
      float alpha = __expf(mrow[r] - nm);
      float p0 = __expf(v0 - nm), p1 = __expf(v1 - nm);
      float rs = p0 + p1;
      #pragma unroll
      for (int msk = 1; msk < 16; msk <<= 1) rs += __shfl_xor(rs, msk, 32);
      lrow[r] = lrow[r] * alpha + rs;
      mrow[r] = nm;
      #pragma unroll
      for (int a = 0; a < 4; ++a) yacc[a][r] = yacc[a][r] * alpha;
      Ps[wave][m][ln]      = f2bf(p0);     // P in row-major LDS (D-layout writes)
      Ps[wave][m][16 + ln] = f2bf(p1);
    }
    asm volatile("s_wait_dscnt 0" ::: "memory");   // DS in-order per wave; fence compiler too

    // ---- P (16x32) as A-fragment, Y += P @ V (V^T rows contiguous) ----
    FragB pf;
    pf.q[0] = *(const uint4*)&Ps[wave][ln][ko];
    pf.q[1] = *(const uint4*)&Ps[wave][ln][16 + ko];
    #pragma unroll
    for (int a = 0; a < 4; ++a) {
      FragB vf;
      const unsigned short* vr = Vp + (size_t)(a * 16 + ln) * Tn + kb;
      vf.q[0] = *(const uint4*)(vr + ko);
      vf.q[1] = *(const uint4*)(vr + 16 + ko);
      yacc[a] = __builtin_amdgcn_wmma_f32_16x16x32_bf16(
          false, pf.v, false, vf.v, (short)0, yacc[a], false, false);
    }
  }

  // ---- normalize and store Y as bf16 [B,T,C] ----
  #pragma unroll
  for (int a = 0; a < 4; ++a) {
    #pragma unroll
    for (int r = 0; r < 8; ++r) {
      int m = r + 8 * hi;
      int q = qb + m;
      int d = a * 16 + ln;
      float inv = 1.0f / lrow[r];
      Yb[((size_t)(b * Tn + q) * Cn) + h * Dn + d] = f2bf(yacc[a][r] * inv);
    }
  }
}

// ---------------------------------------------------------------------------
// GEMM2: out = Yb[4096x1024] @ WpbT [1024][1024] (N-major), fp32 output
// ---------------------------------------------------------------------------
__global__ __launch_bounds__(256) void gemm_proj(const unsigned short* __restrict__ Yb,
                                                 const unsigned short* __restrict__ WpbT,
                                                 float* __restrict__ out) {
  __shared__ __align__(16) unsigned short Bt[2][64][40];
  const int wave = threadIdx.x >> 5, lane = threadIdx.x & 31;
  const int hi = lane >> 4, ln = lane & 15, ko = hi * 8;
  const int mBase = blockIdx.y * 256 + wave * 32;
  const int nBase = blockIdx.x * 64;
  const int sn = threadIdx.x >> 2, sseg = (threadIdx.x & 3) * 8;

  v8f acc[2][4] = {};
  gemm_core(Yb, WpbT, Bt, mBase, nBase, ln, ko, sn, sseg, acc);

  #pragma unroll
  for (int mh = 0; mh < 2; ++mh) {
    #pragma unroll
    for (int nt = 0; nt < 4; ++nt) {
      #pragma unroll
      for (int r = 0; r < 8; ++r) {
        int gm = mBase + mh * 16 + r + 8 * hi;
        int gn = nBase + nt * 16 + ln;
        out[(size_t)gm * Cn + gn] = acc[mh][nt][r];
      }
    }
  }
}

// ---------------------------------------------------------------------------
// host launcher
// ---------------------------------------------------------------------------
extern "C" void kernel_launch(void* const* d_in, const int* in_sizes, int n_in,
                              void* d_out, int out_size, void* d_ws, size_t ws_size,
                              hipStream_t stream) {
  const float* x      = (const float*)d_in[0];   // [B,T,C]
  const float* w_attn = (const float*)d_in[1];   // [C,3C]
  const float* w_proj = (const float*)d_in[2];   // [C,C]
  float* out = (float*)d_out;

  char* ws = (char*)d_ws;
  size_t off = 0;
  unsigned short* Xb   = (unsigned short*)(ws + off); off += (size_t)Mrows * Cn * 2;  // 8 MiB
  unsigned short* WabT = (unsigned short*)(ws + off); off += (size_t)Cn * C3 * 2;     // 6 MiB
  unsigned short* WpbT = (unsigned short*)(ws + off); off += (size_t)Cn * Cn * 2;     // 2 MiB
  unsigned short* Qm   = (unsigned short*)(ws + off); off += (size_t)Mrows * Cn * 2;  // 8 MiB
  unsigned short* Km   = (unsigned short*)(ws + off); off += (size_t)Mrows * Cn * 2;  // 8 MiB
  unsigned short* Vt   = (unsigned short*)(ws + off); off += (size_t)Mrows * Cn * 2;  // 8 MiB
  unsigned short* Yb   = (unsigned short*)(ws + off); off += (size_t)Mrows * Cn * 2;  // 8 MiB

  // fp32 -> bf16 (x stays row-major; weights transposed to N-major)
  cvt_bf16<<<(Mrows * Cn) / (8 * 256), 256, 0, stream>>>(x, Xb, Mrows * Cn);
  cvt_t_bf16<<<dim3(C3 / 32, Cn / 32), 256, 0, stream>>>(w_attn, WabT, Cn, C3);
  cvt_t_bf16<<<dim3(Cn / 32, Cn / 32), 256, 0, stream>>>(w_proj, WpbT, Cn, Cn);

  // qkv GEMM + scatter to [B,H,T,D] / V^T [B,H,D,T]
  gemm_qkv<<<dim3(C3 / 64, Mrows / 256), 256, 0, stream>>>(Xb, WabT, Qm, Km, Vt);

  // flash attention: 4 waves per block, one 16-query tile each
  attn<<<dim3(Bn * Hn * (Tn / 16) / 4), 128, 0, stream>>>(Qm, Km, Vt, Yb);

  // output projection
  gemm_proj<<<dim3(Cn / 64, Mrows / 256), 256, 0, stream>>>(Yb, WpbT, out);
}
